// MultiHeadAttention_18356690223421
// MI455X (gfx1250) — matrix operand order repaired and checked by reading, not compiled, not run on hardware
//
#include <hip/hip_runtime.h>
#include <hip/hip_bf16.h>

// Problem constants (from reference)
constexpr int Bc  = 8;
constexpr int Sc  = 384;
constexpr int Dc  = 1024;
constexpr int Hc  = 16;
constexpr int HDc = 64;        // head dim
constexpr int ST  = Sc / 16;   // 24 tiles along sequence

typedef __bf16 v16bf __attribute__((ext_vector_type(16)));
typedef __bf16 v8bf  __attribute__((ext_vector_type(8)));
typedef __bf16 v4bf  __attribute__((ext_vector_type(4)));
typedef float  v8f   __attribute__((ext_vector_type(8)));

// ---------------------------------------------------------------------------
// WMMA fragment helpers for V_WMMA_F32_16X16X32_BF16 (contiguous k order, same in A and B).
// A (16x32): lane m=lane&15, K = kofs..kofs+15 (elems 0..15),
//            kofs = (lane>=16)*16.
//            -> two contiguous 16B loads per lane from row-major bf16.
// B (32x16): N = lane&15; K along elems; lanes 16-31 K offset +16.
//            From an N-major (transposed) source -> one contiguous 32B span.
// C (16x16 f32): N = lane&15; M = elem + 8*(lane>=16).
// ---------------------------------------------------------------------------

__device__ inline v16bf cat8(v8bf lo, v8bf hi) {
  return __builtin_shufflevector(lo, hi, 0, 1, 2, 3, 4, 5, 6, 7, 8, 9, 10, 11,
                                 12, 13, 14, 15);
}

__device__ inline v16bf ld_A(const __bf16* __restrict__ base, int ld,
                             int lane) {
  int m    = lane & 15;
  int kofs = (lane >> 4) << 4;
  const __bf16* p = base + (size_t)m * ld + kofs;
  v8bf lo = *(const v8bf*)p;
  v8bf hi = *(const v8bf*)(p + 8);
  return cat8(lo, hi);
}

__device__ inline v16bf ld_BT(const __bf16* __restrict__ base, int ld,
                              int lane) {
  int n  = lane & 15;
  int kb = (lane >> 4) << 4;
  const __bf16* p = base + (size_t)n * ld + kb;
  v8bf lo = *(const v8bf*)p;
  v8bf hi = *(const v8bf*)(p + 8);
  return cat8(lo, hi);
}

__device__ inline v8f wmma_bf16(v16bf a, v16bf b, v8f c) {
  return __builtin_amdgcn_wmma_f32_16x16x32_bf16(false, a, false, b,
                                                 (short)0, c, false, false);
}

// ---------------------------------------------------------------------------
// Pack kernels (run once; everything stays L2-resident afterwards)
// ---------------------------------------------------------------------------

__global__ __launch_bounds__(256) void pack_kernel(
    const float* __restrict__ src, __bf16* __restrict__ dst) {
  int idx = blockIdx.x * 256 + threadIdx.x;
  const float4 f = ((const float4*)src)[idx];
  v4bf o;
  o[0] = (__bf16)f.x; o[1] = (__bf16)f.y; o[2] = (__bf16)f.z; o[3] = (__bf16)f.w;
  ((v4bf*)dst)[idx] = o;
}

// fp32 (K x N) -> bf16 transposed (N x K); thread per output element
__global__ __launch_bounds__(256) void packT_kernel(
    const float* __restrict__ src, __bf16* __restrict__ dst, int K, int N) {
  int idx = blockIdx.x * 256 + threadIdx.x;
  int n = idx / K;
  int k = idx % K;
  dst[idx] = (__bf16)src[(size_t)k * N + n];
}

// ---------------------------------------------------------------------------
// Kernel 1: QKV projection. 64x64 output per wave (16 accumulators, 1.0
// b128-load per WMMA). Scatters into q (B,H,S,HD), k (B,H,S,HD),
// vT (B,H,HD,S) bf16 buffers.
// ---------------------------------------------------------------------------
__global__ __launch_bounds__(128) void qkv_kernel(
    const __bf16* __restrict__ xb, const __bf16* __restrict__ wT,
    const float* __restrict__ bias, __bf16* __restrict__ qb,
    __bf16* __restrict__ kb, __bf16* __restrict__ vT) {
  int wid  = blockIdx.x * 4 + (threadIdx.x >> 5);
  int lane = threadIdx.x & 31;
  const int NG = (3 * Dc) / 64;  // 48
  int mg = wid / NG;             // 48 m-groups of 4 tiles
  int ng = wid % NG;

  const __bf16* Abase = xb + (size_t)mg * 64 * Dc;
  const __bf16* Bbase = wT + (size_t)ng * 64 * Dc;
  v8f acc[4][4] = {};
  for (int kk = 0; kk < Dc; kk += 32) {
    __builtin_prefetch(Abase + kk + 64, 0, 1);
    __builtin_prefetch(Bbase + kk + 64, 0, 1);
    v16bf bf[4];
#pragma unroll
    for (int j = 0; j < 4; ++j)
      bf[j] = ld_BT(Bbase + (size_t)(j * 16) * Dc + kk, Dc, lane);
#pragma unroll
    for (int i = 0; i < 4; ++i) {
      v16bf a = ld_A(Abase + (size_t)(i * 16) * Dc + kk, Dc, lane);
#pragma unroll
      for (int j = 0; j < 4; ++j) acc[i][j] = wmma_bf16(a, bf[j], acc[i][j]);
    }
  }

  int mb = (lane >> 4) << 3;
#pragma unroll
  for (int j = 0; j < 4; ++j) {
    int n     = ng * 64 + j * 16 + (lane & 15);  // column in [0, 3D)
    int h     = n / (3 * HDc);
    int r     = n % (3 * HDc);
    int which = r / HDc;
    int hdi   = r % HDc;
    float bv  = bias[n];
#pragma unroll
    for (int i = 0; i < 4; ++i) {
#pragma unroll
      for (int rr = 0; rr < 8; ++rr) {
        int row = mg * 64 + i * 16 + mb + rr;  // flattened b*S + s
        int b_  = row / Sc;
        int s_  = row % Sc;
        __bf16 o = (__bf16)(acc[i][j][rr] + bv);
        if (which == 0)
          qb[((((size_t)b_ * Hc + h) * Sc + s_) * HDc) + hdi] = o;
        else if (which == 1)
          kb[((((size_t)b_ * Hc + h) * Sc + s_) * HDc) + hdi] = o;
        else
          vT[((((size_t)b_ * Hc + h) * HDc + hdi) * Sc) + s_] = o;
      }
    }
  }
}

// ---------------------------------------------------------------------------
// Kernel 2: XL relative bias. bias[b,h,i2,j2] = sum_c q[b,h,i2,c] *
// rpem[h,i2,j2,c] with batch as the M dimension (8 of 16 rows). rpem gather
// replicates the reference's RAW reshape (s,s,d) -> (h,s,s,hd).
// ---------------------------------------------------------------------------
__device__ inline v16bf ld_A_qbatch(const __bf16* __restrict__ qb, int h,
                                    int i2, int kk, int lane) {
  int m    = lane & 15;
  int kofs = (lane >> 4) << 4;
  v8bf lo = {}, hi = {};
  if (m < 8) {
    const __bf16* p =
        qb + ((((size_t)m * Hc + h) * Sc + i2) * HDc) + kk + kofs;
    lo = *(const v8bf*)p;
    hi = *(const v8bf*)(p + 8);
  }
  return cat8(lo, hi);
}

__device__ inline v16bf ld_B_rpe(const __bf16* __restrict__ table, int h,
                                 int i2, int jbase, int kk, int lane) {
  int n  = lane & 15;
  int kb = (lane >> 4) << 4;
  unsigned ebase = (unsigned)h * (unsigned)(Sc * Sc * HDc) +
                   (unsigned)i2 * (unsigned)(Sc * HDc) +
                   (unsigned)(jbase + n) * (unsigned)HDc + (unsigned)(kk + kb);
  v16bf b;
#pragma unroll
  for (int t = 0; t < 16; ++t) {
    unsigned e   = ebase + t;
    unsigned i   = e / (unsigned)(Sc * Dc);
    unsigned rem = e % (unsigned)(Sc * Dc);
    unsigned j   = rem / (unsigned)Dc;
    unsigned cc  = rem % (unsigned)Dc;
    int tr = (int)j - (int)i + Sc;  // table row
    b[t] = table[(size_t)tr * Dc + cc];
  }
  return b;
}

__global__ __launch_bounds__(128) void rpe_bias_kernel(
    const __bf16* __restrict__ qb, const __bf16* __restrict__ rpeb,
    float* __restrict__ attn) {
  int wid  = blockIdx.x * 4 + (threadIdx.x >> 5);
  int lane = threadIdx.x & 31;
  int jt  = wid % ST;
  int tmp = wid / ST;
  int i2  = tmp % Sc;
  int h   = tmp / Sc;

  v8f c = {};
#pragma unroll
  for (int kk = 0; kk < HDc; kk += 32) {
    v16bf a = ld_A_qbatch(qb, h, i2, kk, lane);
    v16bf b = ld_B_rpe(rpeb, h, i2, jt * 16, kk, lane);
    c = wmma_bf16(a, b, c);
  }
  int n = jt * 16 + (lane & 15);
  if (lane < 16) {  // M rows 0..7 = batch, rows 8..15 padding
#pragma unroll
    for (int r = 0; r < 8; ++r)
      attn[((((size_t)r * Hc + h) * Sc + i2) * Sc) + n] = c[r];
  }
}

// ---------------------------------------------------------------------------
// Kernel 3: logits: scale * q @ k^T accumulated onto bias in d_out.
// Wave computes 64 queries x 64 keys (16 accumulators), K = 64.
// ---------------------------------------------------------------------------
__global__ __launch_bounds__(128) void logits_kernel(
    const __bf16* __restrict__ qb, const __bf16* __restrict__ kb,
    float* __restrict__ attn) {
  int wid  = blockIdx.x * 4 + (threadIdx.x >> 5);
  int lane = threadIdx.x & 31;
  const int G = ST / 4;  // 6 groups of 64
  int jg = wid % G;
  int ig = (wid / G) % G;
  int h  = (wid / (G * G)) % Hc;
  int b_ = wid / (G * G * Hc);

  const __bf16* qbase = qb + (((size_t)b_ * Hc + h) * Sc + ig * 64) * HDc;
  const __bf16* kbase = kb + (((size_t)b_ * Hc + h) * Sc + jg * 64) * HDc;
  v8f acc[4][4] = {};
#pragma unroll
  for (int kk = 0; kk < HDc; kk += 32) {
    v16bf bf[4];
#pragma unroll
    for (int j = 0; j < 4; ++j)
      bf[j] = ld_BT(kbase + (size_t)(j * 16) * HDc + kk, HDc, lane);
#pragma unroll
    for (int i = 0; i < 4; ++i) {
      v16bf a = ld_A(qbase + (size_t)(i * 16) * HDc + kk, HDc, lane);
#pragma unroll
      for (int j = 0; j < 4; ++j) acc[i][j] = wmma_bf16(a, bf[j], acc[i][j]);
    }
  }
  const float scale = 0.125f;  // 1/sqrt(64)
  float* obase =
      attn + ((((size_t)b_ * Hc + h) * Sc + ig * 64) * Sc) + jg * 64;
  int n  = lane & 15;
  int mb = (lane >> 4) << 3;
#pragma unroll
  for (int i = 0; i < 4; ++i) {
#pragma unroll
    for (int j = 0; j < 4; ++j) {
#pragma unroll
      for (int r = 0; r < 8; ++r) {
        size_t idx = (size_t)(i * 16 + mb + r) * Sc + j * 16 + n;
        obase[idx] = obase[idx] + scale * acc[i][j][r];
      }
    }
  }
}

// ---------------------------------------------------------------------------
// Kernel 4: row softmax (wave per row of 384). fp32 to d_out + bf16 copy.
// ---------------------------------------------------------------------------
__global__ __launch_bounds__(128) void softmax_kernel(
    float* __restrict__ attn, __bf16* __restrict__ attn_bf) {
  int row  = blockIdx.x * 4 + (threadIdx.x >> 5);
  int lane = threadIdx.x & 31;
  float* p = attn + (size_t)row * Sc;

  float v[12];
  float mx = -3.0e38f;
#pragma unroll
  for (int t = 0; t < 12; ++t) {
    v[t] = p[lane + 32 * t];
    mx = fmaxf(mx, v[t]);
  }
#pragma unroll
  for (int off = 16; off >= 1; off >>= 1)
    mx = fmaxf(mx, __shfl_xor(mx, off, 32));

  float sum = 0.0f;
#pragma unroll
  for (int t = 0; t < 12; ++t) {
    v[t] = __expf(v[t] - mx);
    sum += v[t];
  }
#pragma unroll
  for (int off = 16; off >= 1; off >>= 1)
    sum += __shfl_xor(sum, off, 32);

  float inv = 1.0f / sum;
  __bf16* q = attn_bf + (size_t)row * Sc;
#pragma unroll
  for (int t = 0; t < 12; ++t) {
    float o = v[t] * inv;
    p[lane + 32 * t] = o;
    q[lane + 32 * t] = (__bf16)o;
  }
}

// ---------------------------------------------------------------------------
// Kernel 5: values = attention @ v. Wave does 64 rows x 64 cols, K = 384.
// B fragments from transposed V (b,h,hd,S) are contiguous. Output scattered
// to (B,S,D) bf16 for the final projection.
// ---------------------------------------------------------------------------
__global__ __launch_bounds__(128) void values_kernel(
    const __bf16* __restrict__ attn_bf, const __bf16* __restrict__ vT,
    __bf16* __restrict__ values) {
  int wid  = blockIdx.x * 4 + (threadIdx.x >> 5);
  int lane = threadIdx.x & 31;
  const int G = ST / 4;  // 6
  int ig = wid % G;
  int h  = (wid / G) % Hc;
  int b_ = wid / (G * Hc);

  const __bf16* Abase = attn_bf + (((size_t)b_ * Hc + h) * Sc + ig * 64) * Sc;
  const __bf16* Bbase = vT + (((size_t)b_ * Hc + h) * HDc) * Sc;
  v8f acc[4][4] = {};
  for (int kk = 0; kk < Sc; kk += 32) {
    __builtin_prefetch(Abase + kk + 64, 0, 1);
    v16bf bf[4];
#pragma unroll
    for (int j = 0; j < 4; ++j)
      bf[j] = ld_BT(Bbase + (size_t)(j * 16) * Sc + kk, Sc, lane);
#pragma unroll
    for (int i = 0; i < 4; ++i) {
      v16bf a = ld_A(Abase + (size_t)(i * 16) * Sc + kk, Sc, lane);
#pragma unroll
      for (int j = 0; j < 4; ++j) acc[i][j] = wmma_bf16(a, bf[j], acc[i][j]);
    }
  }
  int mb = (lane >> 4) << 3;
#pragma unroll
  for (int j = 0; j < 4; ++j) {
    int col = h * HDc + j * 16 + (lane & 15);
#pragma unroll
    for (int i = 0; i < 4; ++i) {
#pragma unroll
      for (int r = 0; r < 8; ++r) {
        int s_ = ig * 64 + i * 16 + mb + r;
        values[((size_t)b_ * Sc + s_) * Dc + col] = (__bf16)acc[i][j][r];
      }
    }
  }
}

// ---------------------------------------------------------------------------
// Kernel 6: out = values @ out_w + out_b -> fp32 d_out. 64x64 per wave.
// ---------------------------------------------------------------------------
__global__ __launch_bounds__(128) void out_kernel(
    const __bf16* __restrict__ values, const __bf16* __restrict__ owT,
    const float* __restrict__ ob, float* __restrict__ out) {
  int wid  = blockIdx.x * 4 + (threadIdx.x >> 5);
  int lane = threadIdx.x & 31;
  const int NG = Dc / 64;  // 16
  int mg = wid / NG;       // 48 m-groups
  int ng = wid % NG;

  const __bf16* Abase = values + (size_t)mg * 64 * Dc;
  const __bf16* Bbase = owT + (size_t)ng * 64 * Dc;
  v8f acc[4][4] = {};
  for (int kk = 0; kk < Dc; kk += 32) {
    __builtin_prefetch(Abase + kk + 64, 0, 1);
    __builtin_prefetch(Bbase + kk + 64, 0, 1);
    v16bf bf[4];
#pragma unroll
    for (int j = 0; j < 4; ++j)
      bf[j] = ld_BT(Bbase + (size_t)(j * 16) * Dc + kk, Dc, lane);
#pragma unroll
    for (int i = 0; i < 4; ++i) {
      v16bf a = ld_A(Abase + (size_t)(i * 16) * Dc + kk, Dc, lane);
#pragma unroll
      for (int j = 0; j < 4; ++j) acc[i][j] = wmma_bf16(a, bf[j], acc[i][j]);
    }
  }
  int mb = (lane >> 4) << 3;
#pragma unroll
  for (int j = 0; j < 4; ++j) {
    int n = ng * 64 + j * 16 + (lane & 15);
    float bv = ob[n];
#pragma unroll
    for (int i = 0; i < 4; ++i) {
#pragma unroll
      for (int r = 0; r < 8; ++r)
        out[(size_t)(mg * 64 + i * 16 + mb + r) * Dc + n] = acc[i][j][r] + bv;
    }
  }
}

// ---------------------------------------------------------------------------
// Launch
// ---------------------------------------------------------------------------
extern "C" void kernel_launch(void* const* d_in, const int* in_sizes, int n_in,
                              void* d_out, int out_size, void* d_ws,
                              size_t ws_size, hipStream_t stream) {
  (void)in_sizes; (void)n_in; (void)out_size; (void)ws_size;

  const float* x     = (const float*)d_in[0];
  const float* qkv_w = (const float*)d_in[1];
  const float* qkv_b = (const float*)d_in[2];
  const float* out_w = (const float*)d_in[3];
  const float* out_b = (const float*)d_in[4];
  const float* rpe   = (const float*)d_in[5];

  float* out  = (float*)d_out;               // B*S*D fp32
  float* attn = out + (size_t)Bc * Sc * Dc;  // B*H*S*S fp32

  // Workspace layout (bf16, ~80 MB total; L2 is 192 MB)
  const size_t NQ = (size_t)Bc * Hc * Sc * HDc;  // 3,145,728
  __bf16* qb      = (__bf16*)d_ws;
  __bf16* kb      = qb + NQ;
  __bf16* vT      = kb + NQ;                       // (B,H,HD,S)
  __bf16* attn_bf = vT + NQ;                       // B*H*S*S
  __bf16* values  = attn_bf + (size_t)Bc * Hc * Sc * Sc;  // (B,S,D)
  __bf16* xb      = values + NQ;                   // B*S*D bf16
  __bf16* wT      = xb + NQ;                       // (3D, D)
  __bf16* owT     = wT + (size_t)3 * Dc * Dc;      // (D, D)
  __bf16* rpeb    = owT + (size_t)Dc * Dc;         // (2*512-1, D)

  dim3 blk128(128), blk256(256);

  // 0) pack / transposed-pack to bf16
  pack_kernel<<<(int)(NQ / 1024), blk256, 0, stream>>>(x, xb);
  pack_kernel<<<(1023 * Dc) / 1024, blk256, 0, stream>>>(rpe, rpeb);
  packT_kernel<<<(3 * Dc * Dc) / 256, blk256, 0, stream>>>(qkv_w, wT, Dc,
                                                           3 * Dc);
  packT_kernel<<<(Dc * Dc) / 256, blk256, 0, stream>>>(out_w, owT, Dc, Dc);

  // 1) QKV projection: 48 m-groups x 48 n-groups (64x64 per wave)
  qkv_kernel<<<(48 * 48) / 4, blk128, 0, stream>>>(xb, wT, qkv_b, qb, kb, vT);
  // 2) XL relative bias -> attention region (H*S*24 waves)
  rpe_bias_kernel<<<(Hc * Sc * ST) / 4, blk128, 0, stream>>>(qb, rpeb, attn);
  // 3) logits accumulate (B*H*6*6 waves, 64x64 per wave)
  logits_kernel<<<(Bc * Hc * 6 * 6) / 4, blk128, 0, stream>>>(qb, kb, attn);
  // 4) softmax (B*H*S rows)
  softmax_kernel<<<(Bc * Hc * Sc) / 4, blk128, 0, stream>>>(attn, attn_bf);
  // 5) attention @ V (B*H*6 waves, 64x64 per wave)
  values_kernel<<<(Bc * Hc * 6) / 4, blk128, 0, stream>>>(attn_bf, vT, values);
  // 6) output projection (48 x 16 waves, 64x64 per wave)
  out_kernel<<<(48 * 16) / 4, blk128, 0, stream>>>(values, owT, out_b, out);
}
